// GPSA_15298673508692
// MI455X (gfx1250) — matrix operand order repaired and checked by reading, not compiled, not run on hardware
//
#include <hip/hip_runtime.h>
#include <hip/hip_bf16.h>

typedef __attribute__((ext_vector_type(8)))  _Float16 v8h;
typedef __attribute__((ext_vector_type(16))) _Float16 v16h;
typedef __attribute__((ext_vector_type(8)))  float    v8f;
typedef __attribute__((ext_vector_type(4)))  float    v4f;

#define WMMA16(a, b, c) \
  __builtin_amdgcn_wmma_f32_16x16x32_f16(false, (a), false, (b), (short)0, (c), false, false)

static constexpr int Bsz = 16, Ntok = 576, Cdim = 768, Hh = 12, Dh = 64;

__device__ __forceinline__ v16h pack16(v8h lo, v8h hi) {
  v16h r;
#pragma unroll
  for (int i = 0; i < 8; ++i) { r[i] = lo[i]; r[i + 8] = hi[i]; }
  return r;
}

// A fragment (16x32, f16): lane<16 holds row=lane, K {k0+0..7, k0+16..23};
// lane>=16 holds row=lane-16, K {k0+8..15, k0+24..31}.
__device__ __forceinline__ v16h load_a16(const _Float16* __restrict__ rowp, int k0, int hi) {
  v8h lo = *(const v8h*)(rowp + k0 + hi * 8);
  v8h hb = *(const v8h*)(rowp + k0 + hi * 8 + 16);
  return pack16(lo, hb);
}

// B fragment (32x16, f16): lane<16 holds col=lane, K {k0+0..7, k0+16..23};
// lane>=16 holds col=lane-16, K {k0+8..15, k0+24..31}. colp = base of the column (row of B^T).
__device__ __forceinline__ v16h load_b16(const _Float16* __restrict__ colp, int k0, int hi) {
  v8h lo = *(const v8h*)(colp + k0 + hi * 8);
  v8h hb = *(const v8h*)(colp + k0 + hi * 8 + 16);
  return pack16(lo, hb);
}

__device__ __forceinline__ void zero_acc(v8f* acc, int n) {
  for (int c = 0; c < n; ++c)
#pragma unroll
    for (int i = 0; i < 8; ++i) acc[c][i] = 0.0f;
}

// One wave computes a 16(M) x 64(N) strip: acc[4] of 16x16 f32 tiles.
__device__ __forceinline__ void wave_gemm_16x64(
    const _Float16* __restrict__ A, int lda, int row0,
    const _Float16* __restrict__ Bt, int ldb, int col0,
    int K, v8f acc[4]) {
  const int lane = threadIdx.x & 31;
  const int l15 = lane & 15, hi = lane >> 4;
  const _Float16* arow = A + (size_t)(row0 + l15) * lda;
  const _Float16* bcol = Bt + (size_t)(col0 + l15) * ldb;
  for (int k0 = 0; k0 < K; k0 += 32) {
    __builtin_prefetch(arow + k0 + 256, 0, 3);   // global_prefetch_b8
    v16h a = load_a16(arow, k0, hi);
#pragma unroll
    for (int cs = 0; cs < 4; ++cs) {
      v16h b = load_b16(bcol + (size_t)(cs * 16) * ldb, k0, hi);
      acc[cs] = WMMA16(a, b, acc[cs]);
    }
  }
}

// ---------------- conversion kernels ----------------
__global__ void k_cvt(const float* __restrict__ src, _Float16* __restrict__ dst, int n) {
  for (int i = blockIdx.x * blockDim.x + threadIdx.x; i < n; i += gridDim.x * blockDim.x)
    dst[i] = (_Float16)src[i];
}

// src [K][Nc] f32 -> dst [Nc][K] f16 (transpose so B-fragment loads are contiguous)
__global__ void k_cvt_T(const float* __restrict__ src, _Float16* __restrict__ dst,
                        int K, int Nc) {
  int total = K * Nc;
  for (int i = blockIdx.x * blockDim.x + threadIdx.x; i < total; i += gridDim.x * blockDim.x) {
    int k = i / Nc, n = i - k * Nc;
    dst[(size_t)n * K + k] = (_Float16)src[i];
  }
}

// ---------------- positional softmax: pos[H][N][N] ----------------
__global__ void k_pos(const float* __restrict__ wpos, const float* __restrict__ bpos,
                      float* __restrict__ pos) {
  int wave = threadIdx.x >> 5, lane = threadIdx.x & 31;
  int row = blockIdx.x * (blockDim.x >> 5) + wave;     // [0, H*N)
  if (row >= Hh * Ntok) return;
  int h = row / Ntok, n = row - h * Ntok;
  float w0 = wpos[h], w1 = wpos[Hh + h], w2 = wpos[2 * Hh + h], bb = bpos[h];
  int nx = n % 24, ny = n / 24;
  float vals[18];
  float mx = -1e30f;
#pragma unroll
  for (int t = 0; t < 18; ++t) {
    int j = lane + t * 32;
    float dx = (float)((j % 24) - nx);
    float dy = (float)((j / 24) - ny);
    float lg = dx * w0 + dy * w1 + (dx * dx + dy * dy) * w2 + bb;
    vals[t] = lg;
    mx = fmaxf(mx, lg);
  }
  for (int off = 16; off; off >>= 1) mx = fmaxf(mx, __shfl_xor(mx, off, 32));
  float sum = 0.f;
#pragma unroll
  for (int t = 0; t < 18; ++t) { vals[t] = __expf(vals[t] - mx); sum += vals[t]; }
  for (int off = 16; off; off >>= 1) sum += __shfl_xor(sum, off, 32);
  float inv = 1.0f / sum;
  float* prow = pos + (size_t)row * Ntok;
#pragma unroll
  for (int t = 0; t < 18; ++t) prow[lane + t * 32] = vals[t] * inv;
}

// ---------------- qk GEMM: x[9216,768] @ w_qk -> q,k [B,H,N,Dh] f16 ----------------
__global__ void __launch_bounds__(128) k_qk_gemm(
    const _Float16* __restrict__ xh, const _Float16* __restrict__ wqkT,
    _Float16* __restrict__ qh, _Float16* __restrict__ kh) {
  const int Mtiles = (Bsz * Ntok) / 16, Nstrips = (2 * Cdim) / 64;   // 576, 24
  int w = blockIdx.x * (blockDim.x >> 5) + (threadIdx.x >> 5);
  if (w >= Mtiles * Nstrips) return;
  int mt = w / Nstrips, ns = w - mt * Nstrips;
  int row0 = mt * 16, col0 = ns * 64;
  v8f acc[4]; zero_acc(acc, 4);
  wave_gemm_16x64(xh, Cdim, row0, wqkT, Cdim, col0, Cdim, acc);
  int lane = threadIdx.x & 31, l15 = lane & 15, hi = lane >> 4;
#pragma unroll
  for (int cs = 0; cs < 4; ++cs) {
    int gc = col0 + cs * 16 + l15;
    _Float16* dst = (gc >= Cdim) ? kh : qh;
    int rem = gc & (Cdim - 1);
    int h = rem >> 6, d = rem & 63;
#pragma unroll
    for (int r = 0; r < 8; ++r) {
      int gr = row0 + hi * 8 + r;
      int b = gr / Ntok, n = gr - b * Ntok;
      dst[((((size_t)b * Hh + h) * Ntok) + n) * Dh + d] = (_Float16)acc[cs][r];
    }
  }
}

// ---------------- v GEMM: writes f32 v[B,H,N,Dh] (output, NT) + f16 v^T[B,H,Dh,N] ----------------
__global__ void __launch_bounds__(128) k_v_gemm(
    const _Float16* __restrict__ xh, const _Float16* __restrict__ wvT,
    float* __restrict__ vout, _Float16* __restrict__ vTh) {
  const int Mtiles = (Bsz * Ntok) / 16, Nstrips = Cdim / 64;  // 576, 12
  int w = blockIdx.x * (blockDim.x >> 5) + (threadIdx.x >> 5);
  if (w >= Mtiles * Nstrips) return;
  int mt = w / Nstrips, ns = w - mt * Nstrips;
  int row0 = mt * 16, col0 = ns * 64;
  v8f acc[4]; zero_acc(acc, 4);
  wave_gemm_16x64(xh, Cdim, row0, wvT, Cdim, col0, Cdim, acc);
  int lane = threadIdx.x & 31, l15 = lane & 15, hi = lane >> 4;
#pragma unroll
  for (int cs = 0; cs < 4; ++cs) {
    int gc = col0 + cs * 16 + l15;
    int h = gc >> 6, d = gc & 63;
#pragma unroll
    for (int r = 0; r < 8; ++r) {
      int gr = row0 + hi * 8 + r;
      int b = gr / Ntok, n = gr - b * Ntok;
      float val = acc[cs][r];
      size_t bh = (size_t)b * Hh + h;
      __builtin_nontemporal_store(val, &vout[(bh * Ntok + n) * Dh + d]);  // write-once output
      vTh[(bh * Dh + d) * Ntok + n] = (_Float16)val;                      // re-read: keep cached
    }
  }
}

// ---------------- fused attention: scores -> softmax+pos blend -> attn out -> attn@v ----------------
__global__ void __launch_bounds__(32) k_attn(
    const _Float16* __restrict__ qh, const _Float16* __restrict__ kh,
    const _Float16* __restrict__ vTh, const float* __restrict__ pos,
    const float* __restrict__ gating,
    float* __restrict__ attn_out, _Float16* __restrict__ ctxh) {
  __shared__ float s[16][592];                         // 16 queries x 576 keys (+pad)
  int bid = blockIdx.x;
  int qt = bid % 36; int h = (bid / 36) % Hh; int b = bid / (36 * Hh);
  int lane = threadIdx.x & 31, l15 = lane & 15, hi = lane >> 4;
  int q0 = qt * 16;

  // q fragments for full Dh=64 (two K=32 chunks)
  const _Float16* qbase = qh + ((((size_t)b * Hh + h) * Ntok) + q0 + l15) * Dh;
  v16h qa0 = load_a16(qbase, 0, hi);
  v16h qa1 = load_a16(qbase, 32, hi);

  // phase 1: S = (q k^T) * scale, staged to LDS
  const _Float16* kbh = kh + (((size_t)b * Hh + h) * Ntok) * Dh;
  for (int kt = 0; kt < 36; ++kt) {
    const _Float16* kcol = kbh + (size_t)(kt * 16 + l15) * Dh;
    __builtin_prefetch(kcol + 16 * Dh, 0, 3);
    v16h b0 = load_b16(kcol, 0, hi);
    v16h b1 = load_b16(kcol, 32, hi);
    v8f acc; zero_acc(&acc, 1);
    acc = WMMA16(qa0, b0, acc);
    acc = WMMA16(qa1, b1, acc);
#pragma unroll
    for (int r = 0; r < 8; ++r) s[hi * 8 + r][kt * 16 + l15] = acc[r] * 0.125f;
  }
  __syncthreads();

  // phase 2: softmax + gated pos blend. Lane pair per row: row = l15, half = hi
  // (288 floats each, float4-vectorized); halves combined via shfl_xor(16).
  float g = 1.0f / (1.0f + __expf(-gating[h]));
  float invr;                                          // 1/rowsum, needed in phase 3
  {
    float* srow = s[l15];
    const float* prow = pos + (((size_t)h * Ntok) + q0 + l15) * Ntok;
    float* arow = attn_out + ((((size_t)b * Hh + h) * Ntok) + q0 + l15) * Ntok;
    const int base = hi * 288;

    float mx = -1e30f;
    for (int i = 0; i < 288; i += 4) {
      v4f v = *(const v4f*)(srow + base + i);
      mx = fmaxf(mx, fmaxf(fmaxf(v.x, v.y), fmaxf(v.z, v.w)));
    }
    mx = fmaxf(mx, __shfl_xor(mx, 16, 32));

    float sum = 0.f;
    for (int i = 0; i < 288; i += 4) {
      v4f v = *(const v4f*)(srow + base + i);
      v4f e; e.x = __expf(v.x - mx); e.y = __expf(v.y - mx);
      e.z = __expf(v.z - mx); e.w = __expf(v.w - mx);
      *(v4f*)(srow + base + i) = e;
      sum += (e.x + e.y) + (e.z + e.w);
    }
    sum += __shfl_xor(sum, 16, 32);
    float invs = (1.0f - g) / sum;

    float rs = 0.f;
    for (int i = 0; i < 288; i += 4) {
      v4f e = *(const v4f*)(srow + base + i);
      v4f p = *(const v4f*)(prow + base + i);
      v4f a; a.x = e.x * invs + g * p.x; a.y = e.y * invs + g * p.y;
      a.z = e.z * invs + g * p.z; a.w = e.w * invs + g * p.w;
      *(v4f*)(srow + base + i) = a;                    // unnormalized blend kept in LDS
      rs += (a.x + a.y) + (a.z + a.w);
    }
    rs += __shfl_xor(rs, 16, 32);
    invr = 1.0f / rs;

    // write-once attn output: normalized, non-temporal (never re-read from HBM)
    for (int i = 0; i < 288; i += 4) {
      v4f a = *(const v4f*)(srow + base + i);
      a.x *= invr; a.y *= invr; a.z *= invr; a.w *= invr;
      __builtin_nontemporal_store(a, (v4f*)(arow + base + i));
    }
  }
  __syncthreads();

  // phase 3: O = attn @ v via v^T columns (renorm 1/rs folded into f16 A-fragment);
  // ctx [B,N,H,Dh] f16
  const _Float16* vtb = vTh + (((size_t)b * Hh + h) * Dh) * Ntok;
  v8f acc4[4]; zero_acc(acc4, 4);
  for (int kt = 0; kt < 18; ++kt) {
    int kb = kt * 32;
    v16h a;
    {
      const float* srow = s[l15];                      // A row = l15; invr matches this row
      int o0 = kb + hi * 8;
#pragma unroll
      for (int i = 0; i < 8; ++i) {
        a[i]     = (_Float16)(srow[o0 + i] * invr);
        a[i + 8] = (_Float16)(srow[o0 + 16 + i] * invr);
      }
    }
#pragma unroll
    for (int cs = 0; cs < 4; ++cs) {
      const _Float16* vcol = vtb + (size_t)(cs * 16 + l15) * Ntok;
      v16h bb = load_b16(vcol, kb, hi);
      acc4[cs] = WMMA16(a, bb, acc4[cs]);
    }
  }
#pragma unroll
  for (int cs = 0; cs < 4; ++cs) {
    int d = cs * 16 + l15;
#pragma unroll
    for (int r = 0; r < 8; ++r) {
      int n = q0 + hi * 8 + r;
      ctxh[(((size_t)b * Ntok + n) * Hh + h) * Dh + d] = (_Float16)acc4[cs][r];
    }
  }
}

// ---------------- output projection: ctx[9216,768] @ w_proj + b_proj -> f32 out (NT) ----------------
__global__ void __launch_bounds__(128) k_proj_gemm(
    const _Float16* __restrict__ ctxh, const _Float16* __restrict__ wpT,
    const float* __restrict__ bproj, float* __restrict__ out) {
  const int Mtiles = (Bsz * Ntok) / 16, Nstrips = Cdim / 64;
  int w = blockIdx.x * (blockDim.x >> 5) + (threadIdx.x >> 5);
  if (w >= Mtiles * Nstrips) return;
  int mt = w / Nstrips, ns = w - mt * Nstrips;
  int row0 = mt * 16, col0 = ns * 64;
  v8f acc[4]; zero_acc(acc, 4);
  wave_gemm_16x64(ctxh, Cdim, row0, wpT, Cdim, col0, Cdim, acc);
  int lane = threadIdx.x & 31, l15 = lane & 15, hi = lane >> 4;
#pragma unroll
  for (int cs = 0; cs < 4; ++cs) {
    int gc = col0 + cs * 16 + l15;
    float bias = bproj[gc];
#pragma unroll
    for (int r = 0; r < 8; ++r) {
      int gr = row0 + hi * 8 + r;
      __builtin_nontemporal_store(acc[cs][r] + bias, &out[(size_t)gr * Cdim + gc]);
    }
  }
}

extern "C" void kernel_launch(void* const* d_in, const int* in_sizes, int n_in,
                              void* d_out, int out_size, void* d_ws, size_t ws_size,
                              hipStream_t stream) {
  const float* x      = (const float*)d_in[0];
  const float* w_qk   = (const float*)d_in[1];
  const float* w_v    = (const float*)d_in[2];
  const float* w_proj = (const float*)d_in[3];
  const float* b_proj = (const float*)d_in[4];
  const float* w_pos  = (const float*)d_in[5];
  const float* b_pos  = (const float*)d_in[6];
  const float* gating = (const float*)d_in[7];

  // output sections: out | attn | v
  float* out  = (float*)d_out;
  float* attn = out + (size_t)Bsz * Ntok * Cdim;
  float* vout = attn + (size_t)Bsz * Hh * Ntok * Ntok;

  // workspace layout (f16 staging + pos table), ~91 MB
  char* ws = (char*)d_ws;
  size_t o = 0;
  _Float16* xh     = (_Float16*)(ws + o); o += (size_t)Bsz * Ntok * Cdim * 2;
  _Float16* wqkT   = (_Float16*)(ws + o); o += (size_t)2 * Cdim * Cdim * 2;
  _Float16* wvT    = (_Float16*)(ws + o); o += (size_t)Cdim * Cdim * 2;
  _Float16* wpT    = (_Float16*)(ws + o); o += (size_t)Cdim * Cdim * 2;
  _Float16* qh     = (_Float16*)(ws + o); o += (size_t)Bsz * Hh * Ntok * Dh * 2;
  _Float16* kh     = (_Float16*)(ws + o); o += (size_t)Bsz * Hh * Ntok * Dh * 2;
  _Float16* vTh    = (_Float16*)(ws + o); o += (size_t)Bsz * Hh * Dh * Ntok * 2;
  _Float16* ctxh   = (_Float16*)(ws + o); o += (size_t)Bsz * Ntok * Cdim * 2;
  float*    pos    = (float*)(ws + o);    o += (size_t)Hh * Ntok * Ntok * 4;

  // 1) stage f16 operands
  k_cvt<<<4096, 256, 0, stream>>>(x, xh, Bsz * Ntok * Cdim);
  k_cvt_T<<<(2 * Cdim * Cdim + 255) / 256, 256, 0, stream>>>(w_qk, wqkT, Cdim, 2 * Cdim);
  k_cvt_T<<<(Cdim * Cdim + 255) / 256, 256, 0, stream>>>(w_v, wvT, Cdim, Cdim);
  k_cvt_T<<<(Cdim * Cdim + 255) / 256, 256, 0, stream>>>(w_proj, wpT, Cdim, Cdim);

  // 2) positional softmax table [H,N,N]
  k_pos<<<(Hh * Ntok + 3) / 4, 128, 0, stream>>>(w_pos, b_pos, pos);

  // 3) projections
  k_qk_gemm<<<(576 * 24 + 3) / 4, 128, 0, stream>>>(xh, wqkT, qh, kh);
  k_v_gemm<<<(576 * 12 + 3) / 4, 128, 0, stream>>>(xh, wvT, vout, vTh);

  // 4) fused attention (scores, softmax, gated pos blend, renorm, attn@v)
  k_attn<<<Bsz * Hh * 36, 32, 0, stream>>>(qh, kh, vTh, pos, gating, attn, ctxh);

  // 5) output projection + bias
  k_proj_gemm<<<(576 * 12 + 3) / 4, 128, 0, stream>>>(ctxh, wpT, b_proj, out);
}
